// NSA_12859132084901
// MI455X (gfx1250) — matrix operand order repaired and checked by reading, not compiled, not run on hardware
//
#include <hip/hip_runtime.h>
#include <stdint.h>

#define T_SZ 2048
#define HIDD 2048
#define NH   32
#define NG   2
#define HGQ  16
#define DQK  64
#define DVV  64
#define BLKB 32
#define NBB  64
#define NSELK 8
#define WINW 256
#define OUTD 2400
#define OUTD_PAD 2432          // OUTD rounded up to 128
#define SCALE_F 0.125f
#define NEGF (-1e30f)

typedef __bf16 bf16_t;
typedef __attribute__((ext_vector_type(16))) __bf16 v16bf;
typedef __attribute__((ext_vector_type(8)))  __bf16 v8bf;
typedef __attribute__((ext_vector_type(8)))  float  v8f;
typedef int v4i_t __attribute__((vector_size(16)));

#define AS1 __attribute__((address_space(1)))
#define AS3 __attribute__((address_space(3)))

#if __has_builtin(__builtin_amdgcn_global_load_async_to_lds_b128)
#define HAVE_ASYNC_LDS 1
#else
#define HAVE_ASYNC_LDS 0
#endif

#if HAVE_ASYNC_LDS
static __device__ inline void async_cp16(const bf16_t* g, bf16_t* l) {
  // 16 bytes per lane, global -> LDS, tracked by ASYNCcnt
  __builtin_amdgcn_global_load_async_to_lds_b128((AS1 v4i_t*)g, (AS3 v4i_t*)l, 0, 0);
}
static __device__ inline void wait_async_all() {
#if __has_builtin(__builtin_amdgcn_s_wait_asynccnt)
  __builtin_amdgcn_s_wait_asynccnt(0);
#else
  asm volatile("s_wait_asynccnt 0x0" ::: "memory");
#endif
}
#endif

static __device__ inline v8f wmma_bf16(v16bf a, v16bf b, v8f c) {
  return __builtin_amdgcn_wmma_f32_16x16x32_bf16(false, a, false, b, (short)0, c, false, false);
}

// A-fragment: lane holds 16 contiguous K; caller passes p = row + 16*(lane>>4)
static __device__ inline v16bf frag_a_ptr(const bf16_t* p) {
  v8bf lo = *(const v8bf*)(p);
  v8bf hi = *(const v8bf*)(p + 8);
  v16bf r;
#pragma unroll
  for (int i = 0; i < 8; ++i) { r[i] = lo[i]; r[i + 8] = hi[i]; }
  return r;
}

// B-fragment: lane holds 16 contiguous K; caller passes p = col_row + 16*(lane>>4)
static __device__ inline v16bf frag_b_ptr(const bf16_t* p) {
  v8bf lo = *(const v8bf*)(p);
  v8bf hi = *(const v8bf*)(p + 8);
  v16bf r;
#pragma unroll
  for (int i = 0; i < 8; ++i) { r[i] = lo[i]; r[i + 8] = hi[i]; }
  return r;
}

static __device__ inline float red_max16(float v) {
#pragma unroll
  for (int m = 1; m <= 8; m <<= 1) v = fmaxf(v, __shfl_xor(v, m, 32));
  return v;
}
static __device__ inline float red_sum16(float v) {
#pragma unroll
  for (int m = 1; m <= 8; m <<= 1) v += __shfl_xor(v, m, 32);
  return v;
}

// ---------------- pre-passes: convert / transpose to bf16 ----------------
__global__ __launch_bounds__(256)
void cvt_bf16_kernel(const float* __restrict__ in, bf16_t* __restrict__ out, int n) {
  int i = blockIdx.x * 256 + threadIdx.x;
  if (i < n) out[i] = (bf16_t)in[i];
}

// in: f32 [K][N] row-major -> out: bf16 [Npad][K] row-major (rows >= N zero-filled)
__global__ __launch_bounds__(256)
void transpose_bf16_kernel(const float* __restrict__ in, bf16_t* __restrict__ out,
                           int K, int N, int Npad) {
  int idx = blockIdx.x * 256 + threadIdx.x;
  if (idx >= Npad * K) return;
  int n = idx / K, k = idx - n * K;
  float v = (n < N) ? in[(size_t)k * N + n] : 0.f;
  out[idx] = (bf16_t)v;
}

// ---------------- GEMM: C[M,N] = A[M,K] @ Bt[N,K]^T  (bf16 in, f32 out) ----------------
// A row-major [M][K] bf16; Bt row-major [Npad][K] bf16 (pre-transposed weights).
// Double-buffered LDS tiles filled with async global->LDS copies when available.
#define BM 128
#define BN 128
#define BK 32

__global__ __launch_bounds__(256)
void gemm_bf16_kernel(const bf16_t* __restrict__ A, const bf16_t* __restrict__ Bt,
                      float* __restrict__ C, int M, int N, int K) {
  __shared__ bf16_t Asd[2][BM][BK];
  __shared__ bf16_t Bsd[2][BN][BK];
  const int tid = threadIdx.x;
  const int l = tid & 31, lr = l & 15, lh = l >> 4;
  const int wid = tid >> 5;
  const int wm = (wid & 1) * 64;
  const int wn = (wid >> 1) * 32;
  const int m0 = blockIdx.y * BM;
  const int n0 = blockIdx.x * BN;

  // each tile = 512 chunks of 16B; 256 threads -> 2 chunks each
  auto stage = [&](int buf, int k0) {
#pragma unroll
    for (int cc = 0; cc < 2; ++cc) {
      int c = tid + cc * 256;
      int r = c >> 2, co = (c & 3) * 8;
      const bf16_t* sa = A  + (size_t)(m0 + r) * K + k0 + co;
      const bf16_t* sb = Bt + (size_t)(n0 + r) * K + k0 + co;
#if HAVE_ASYNC_LDS
      async_cp16(sa, &Asd[buf][r][co]);
      async_cp16(sb, &Bsd[buf][r][co]);
#else
      *(v8bf*)&Asd[buf][r][co] = *(const v8bf*)sa;
      *(v8bf*)&Bsd[buf][r][co] = *(const v8bf*)sb;
      if (k0 + BK < K) {
        __builtin_prefetch(sa + BK, 0, 1);
        __builtin_prefetch(sb + BK, 0, 1);
      }
#endif
    }
  };

  v8f acc[4][2];
#pragma unroll
  for (int a = 0; a < 4; ++a)
#pragma unroll
    for (int b = 0; b < 2; ++b)
#pragma unroll
      for (int i = 0; i < 8; ++i) acc[a][b][i] = 0.f;

  const int nk = K / BK;
  stage(0, 0);
  for (int kk = 0; kk < nk; ++kk) {
    const int buf = kk & 1;
#if HAVE_ASYNC_LDS
    wait_async_all();          // my async copies from previous stage() landed
#endif
    __syncthreads();           // everyone's copies landed; prev compute done
    if (kk + 1 < nk) stage(buf ^ 1, (kk + 1) * BK);
#pragma unroll
    for (int mt = 0; mt < 4; ++mt) {
      v16bf a = frag_a_ptr(&Asd[buf][wm + mt * 16 + lr][16 * lh]);
#pragma unroll
      for (int nt = 0; nt < 2; ++nt) {
        v16bf b = frag_b_ptr(&Bsd[buf][wn + nt * 16 + lr][16 * lh]);
        acc[mt][nt] = wmma_bf16(a, b, acc[mt][nt]);
      }
    }
  }
#pragma unroll
  for (int mt = 0; mt < 4; ++mt)
#pragma unroll
    for (int nt = 0; nt < 2; ++nt)
#pragma unroll
      for (int i = 0; i < 8; ++i) {
        int row = m0 + wm + mt * 16 + i + 8 * lh;
        int col = n0 + wn + nt * 16 + lr;
        if (col < N) C[(size_t)row * N + col] = acc[mt][nt][i];
      }
}

// ---------------- RoPE + pack to bf16 (q,k row-major; v transposed), gates ----------------
__global__ __launch_bounds__(256)
void rope_pack_kernel(const float* __restrict__ qkvw, const float* __restrict__ cosb,
                      const float* __restrict__ sinb, bf16_t* __restrict__ qb,
                      bf16_t* __restrict__ kb, bf16_t* __restrict__ vt,
                      float* __restrict__ gates) {
  const int t = blockIdx.x;
  const int tid = threadIdx.x;
  const float* row = qkvw + (size_t)t * OUTD;
  for (int i = tid; i < NH * DQK; i += 256) {
    int d = i & 63;
    float x = row[i];
    float rot = (d < 32) ? -row[i + 32] : row[i - 32];
    float o = x * cosb[t * 64 + d] + rot * sinb[t * 64 + d];
    qb[(size_t)t * (NH * DQK) + i] = (bf16_t)o;
  }
  if (tid < NG * DQK) {                  // k + rope
    int d = tid & 63;
    float x = row[2048 + tid];
    float rot = (d < 32) ? -row[2048 + tid + 32] : row[2048 + tid - 32];
    float o = x * cosb[t * 64 + d] + rot * sinb[t * 64 + d];
    kb[(size_t)t * (NG * DQK) + tid] = (bf16_t)o;
  } else if (tid < 256) {                // v transposed: vt[(g*64+dv)*T + t]
    int j = tid - 128;
    vt[(size_t)j * T_SZ + t] = (bf16_t)row[2176 + j];
  }
  if (tid < NH * 3) {
    float x = row[2304 + tid];
    gates[(size_t)t * (NH * 3) + tid] = 1.f / (1.f + __expf(-x));
  }
}

// ---------------- mean-pool K/V blocks for compressed branch ----------------
__global__ __launch_bounds__(64)
void pool_kernel(const bf16_t* __restrict__ kb, const float* __restrict__ qkvw,
                 bf16_t* __restrict__ kcmp, bf16_t* __restrict__ vtcmp) {
  const int nb = blockIdx.x >> 1, g = blockIdx.x & 1;
  const int d = threadIdx.x;
  float sk = 0.f, sv = 0.f;
  for (int j = 0; j < BLKB; ++j) {
    int t = nb * BLKB + j;
    sk += (float)kb[((size_t)t * NG + g) * DQK + d];
    sv += qkvw[(size_t)t * OUTD + 2176 + g * DVV + d];
  }
  kcmp[((size_t)nb * NG + g) * DQK + d] = (bf16_t)(sk * (1.f / 32.f));
  vtcmp[((size_t)(g * DVV + d)) * NBB + nb] = (bf16_t)(sv * (1.f / 32.f));
}

// ---------------- compressed-branch attention (also emits importance) ----------------
__global__ __launch_bounds__(32)
void cmp_attn_kernel(const bf16_t* __restrict__ qb, const bf16_t* __restrict__ kcmp,
                     const bf16_t* __restrict__ vtcmp, const float* __restrict__ gates,
                     float* __restrict__ ocomb, float* __restrict__ imp) {
  __shared__ bf16_t Pl[16 * 64];
  const int bid = blockIdx.x;
  const int t = bid >> 1, g = bid & 1;
  const int l = threadIdx.x, lr = l & 15, lh = l >> 4;
  const int nvis = (t + 1) >> 5;

  if (nvis == 0) {   // no fully-past block: o_cmp = 0, imp = 0
    for (int i = l; i < HGQ * DVV; i += 32) {
      int h = g * HGQ + (i >> 6);
      ocomb[((size_t)t * NH + h) * DVV + (i & 63)] = 0.f;
    }
    for (int nb = l; nb < NBB; nb += 32) imp[((size_t)t * NG + g) * NBB + nb] = 0.f;
    return;
  }

  const bf16_t* qrow = qb + ((size_t)t * NH + g * HGQ + lr) * DQK;
  v16bf a0 = frag_a_ptr(qrow + 16 * lh);
  v16bf a1 = frag_a_ptr(qrow + 32 + 16 * lh);

  float sacc[4][8];
#pragma unroll
  for (int nt = 0; nt < 4; ++nt) {
    int nb = nt * 16 + lr;
    const bf16_t* krow = kcmp + ((size_t)nb * NG + g) * DQK;
    v16bf b0 = frag_b_ptr(krow + 16 * lh);
    v16bf b1 = frag_b_ptr(krow + 32 + 16 * lh);
    v8f acc;
#pragma unroll
    for (int i = 0; i < 8; ++i) acc[i] = 0.f;
    acc = wmma_bf16(a0, b0, acc);
    acc = wmma_bf16(a1, b1, acc);
    bool vis = (nb < nvis);
#pragma unroll
    for (int i = 0; i < 8; ++i) sacc[nt][i] = vis ? acc[i] * SCALE_F : NEGF;
  }
#pragma unroll
  for (int i = 0; i < 8; ++i) {
    float m = NEGF;
#pragma unroll
    for (int nt = 0; nt < 4; ++nt) m = fmaxf(m, sacc[nt][i]);
    m = red_max16(m);
    float s = 0.f;
#pragma unroll
    for (int nt = 0; nt < 4; ++nt) { float p = __expf(sacc[nt][i] - m); sacc[nt][i] = p; s += p; }
    s = red_sum16(s);
    float inv = 1.f / s;
#pragma unroll
    for (int nt = 0; nt < 4; ++nt)
      Pl[(i + 8 * lh) * 64 + nt * 16 + lr] = (bf16_t)(sacc[nt][i] * inv);
  }
  __syncthreads();
#pragma unroll
  for (int rep = 0; rep < 2; ++rep) {     // importance = sum over heads
    int col = l + rep * 32;
    float s = 0.f;
#pragma unroll
    for (int r = 0; r < 16; ++r) s += (float)Pl[r * 64 + col];
    imp[((size_t)t * NG + g) * NBB + col] = s;
  }
  v8f oacc[4];
#pragma unroll
  for (int dt = 0; dt < 4; ++dt)
#pragma unroll
    for (int i = 0; i < 8; ++i) oacc[dt][i] = 0.f;
#pragma unroll
  for (int c = 0; c < 2; ++c) {
    v16bf pa = frag_a_ptr(Pl + lr * 64 + c * 32 + 16 * lh);
#pragma unroll
    for (int dt = 0; dt < 4; ++dt) {
      int dv = dt * 16 + lr;
      v16bf vb = frag_b_ptr(vtcmp + ((size_t)(g * DVV + dv)) * NBB + c * 32 + 16 * lh);
      oacc[dt] = wmma_bf16(pa, vb, oacc[dt]);
    }
  }
  float grow[8];
#pragma unroll
  for (int i = 0; i < 8; ++i)
    grow[i] = gates[((size_t)t * NH + g * HGQ + i + 8 * lh) * 3 + 0];
#pragma unroll
  for (int dt = 0; dt < 4; ++dt)
#pragma unroll
    for (int i = 0; i < 8; ++i) {
      int h = g * HGQ + i + 8 * lh;
      ocomb[((size_t)t * NH + h) * DVV + dt * 16 + lr] = grow[i] * oacc[dt][i];
    }
}

// ---------------- top-k block selection ----------------
__global__ __launch_bounds__(256)
void topk_kernel(const float* __restrict__ imp, int* __restrict__ sel) {
  int idx = blockIdx.x * 256 + threadIdx.x;
  if (idx >= T_SZ * NG) return;
  int t = idx >> 1;
  const float* ip = imp + (size_t)idx * NBB;
  int cur = t >> 5;
  float sc[NBB];
  for (int nb = 0; nb < NBB; ++nb) {
    if (nb * BLKB <= t) {
      float f = ((nb == 0) ? 1.f : 0.f) + ((nb == cur) ? 1.f : 0.f);
      sc[nb] = ip[nb] + 1e4f * f;
    } else sc[nb] = NEGF;
  }
  unsigned long long used = 0ull;
  for (int s = 0; s < NSELK; ++s) {
    float best = -3.4e38f; int bi = 0;
    for (int nb = 0; nb < NBB; ++nb)
      if (!((used >> nb) & 1ull) && sc[nb] > best) { best = sc[nb]; bi = nb; }
    used |= (1ull << bi);
    sel[(size_t)idx * NSELK + s] = bi;
  }
}

// ---------------- selected-blocks attention ----------------
__global__ __launch_bounds__(32)
void sel_attn_kernel(const bf16_t* __restrict__ qb, const bf16_t* __restrict__ kb,
                     const bf16_t* __restrict__ vt, const int* __restrict__ sel,
                     const float* __restrict__ gates, float* __restrict__ ocomb) {
  __shared__ bf16_t Pl[16 * 256];
  const int bid = blockIdx.x;
  const int t = bid >> 1, g = bid & 1;
  const int l = threadIdx.x, lr = l & 15, lh = l >> 4;

  int sblk[NSELK];
#pragma unroll
  for (int c = 0; c < NSELK; ++c) sblk[c] = sel[(size_t)bid * NSELK + c];

  const bf16_t* qrow = qb + ((size_t)t * NH + g * HGQ + lr) * DQK;
  v16bf a0 = frag_a_ptr(qrow + 16 * lh);
  v16bf a1 = frag_a_ptr(qrow + 32 + 16 * lh);

  float sacc[16][8];
#pragma unroll
  for (int nt = 0; nt < 16; ++nt) {
    int key = sblk[nt >> 1] * BLKB + (nt & 1) * 16 + lr;
    const bf16_t* krow = kb + ((size_t)key * NG + g) * DQK;
    v16bf b0 = frag_b_ptr(krow + 16 * lh);
    v16bf b1 = frag_b_ptr(krow + 32 + 16 * lh);
    v8f acc;
#pragma unroll
    for (int i = 0; i < 8; ++i) acc[i] = 0.f;
    acc = wmma_bf16(a0, b0, acc);
    acc = wmma_bf16(a1, b1, acc);
    bool vis = (key <= t);
#pragma unroll
    for (int i = 0; i < 8; ++i) sacc[nt][i] = vis ? acc[i] * SCALE_F : NEGF;
  }
#pragma unroll
  for (int i = 0; i < 8; ++i) {
    float m = NEGF;
#pragma unroll
    for (int nt = 0; nt < 16; ++nt) m = fmaxf(m, sacc[nt][i]);
    m = red_max16(m);
    float s = 0.f;
#pragma unroll
    for (int nt = 0; nt < 16; ++nt) { float p = __expf(sacc[nt][i] - m); sacc[nt][i] = p; s += p; }
    s = red_sum16(s);
    float inv = 1.f / s;
#pragma unroll
    for (int nt = 0; nt < 16; ++nt)
      Pl[(i + 8 * lh) * 256 + nt * 16 + lr] = (bf16_t)(sacc[nt][i] * inv);
  }
  __syncthreads();
  v8f oacc[4];
#pragma unroll
  for (int dt = 0; dt < 4; ++dt)
#pragma unroll
    for (int i = 0; i < 8; ++i) oacc[dt][i] = 0.f;
#pragma unroll
  for (int c = 0; c < 8; ++c) {
    v16bf pa = frag_a_ptr(Pl + lr * 256 + c * 32 + 16 * lh);
    int kb0 = sblk[c] * BLKB + 16 * lh;
#pragma unroll
    for (int dt = 0; dt < 4; ++dt) {
      int dv = dt * 16 + lr;
      v16bf vb = frag_b_ptr(vt + ((size_t)(g * DVV + dv)) * T_SZ + kb0);
      oacc[dt] = wmma_bf16(pa, vb, oacc[dt]);
    }
  }
  float grow[8];
#pragma unroll
  for (int i = 0; i < 8; ++i)
    grow[i] = gates[((size_t)t * NH + g * HGQ + i + 8 * lh) * 3 + 1];
#pragma unroll
  for (int dt = 0; dt < 4; ++dt)
#pragma unroll
    for (int i = 0; i < 8; ++i) {
      int h = g * HGQ + i + 8 * lh;
      ocomb[((size_t)t * NH + h) * DVV + dt * 16 + lr] += grow[i] * oacc[dt][i];
    }
}

// ---------------- sliding-window attention ----------------
__global__ __launch_bounds__(32)
void win_attn_kernel(const bf16_t* __restrict__ qb, const bf16_t* __restrict__ kb,
                     const bf16_t* __restrict__ vt, const float* __restrict__ gates,
                     float* __restrict__ ocomb) {
  __shared__ bf16_t Pl[16 * 256];
  const int bid = blockIdx.x;
  const int t = bid >> 1, g = bid & 1;
  const int l = threadIdx.x, lr = l & 15, lh = l >> 4;
  const int start = t - (WINW - 1);

  const bf16_t* qrow = qb + ((size_t)t * NH + g * HGQ + lr) * DQK;
  v16bf a0 = frag_a_ptr(qrow + 16 * lh);
  v16bf a1 = frag_a_ptr(qrow + 32 + 16 * lh);

  float sacc[16][8];
#pragma unroll
  for (int nt = 0; nt < 16; ++nt) {
    int pos = start + nt * 16 + lr;
    int key = pos < 0 ? 0 : pos;
    const bf16_t* krow = kb + ((size_t)key * NG + g) * DQK;
    v16bf b0 = frag_b_ptr(krow + 16 * lh);
    v16bf b1 = frag_b_ptr(krow + 32 + 16 * lh);
    v8f acc;
#pragma unroll
    for (int i = 0; i < 8; ++i) acc[i] = 0.f;
    acc = wmma_bf16(a0, b0, acc);
    acc = wmma_bf16(a1, b1, acc);
    bool vis = (pos >= 0);
#pragma unroll
    for (int i = 0; i < 8; ++i) sacc[nt][i] = vis ? acc[i] * SCALE_F : NEGF;
  }
#pragma unroll
  for (int i = 0; i < 8; ++i) {
    float m = NEGF;
#pragma unroll
    for (int nt = 0; nt < 16; ++nt) m = fmaxf(m, sacc[nt][i]);
    m = red_max16(m);
    float s = 0.f;
#pragma unroll
    for (int nt = 0; nt < 16; ++nt) { float p = __expf(sacc[nt][i] - m); sacc[nt][i] = p; s += p; }
    s = red_sum16(s);
    float inv = 1.f / s;
#pragma unroll
    for (int nt = 0; nt < 16; ++nt)
      Pl[(i + 8 * lh) * 256 + nt * 16 + lr] = (bf16_t)(sacc[nt][i] * inv);
  }
  __syncthreads();
  v8f oacc[4];
#pragma unroll
  for (int dt = 0; dt < 4; ++dt)
#pragma unroll
    for (int i = 0; i < 8; ++i) oacc[dt][i] = 0.f;
#pragma unroll
  for (int c = 0; c < 8; ++c) {
    v16bf pa = frag_a_ptr(Pl + lr * 256 + c * 32 + 16 * lh);
    int kbase = start + c * 32 + 16 * lh;
#pragma unroll
    for (int dt = 0; dt < 4; ++dt) {
      int dv = dt * 16 + lr;
      const bf16_t* vrow = vt + ((size_t)(g * DVV + dv)) * T_SZ;
      v16bf vb;   // window base is arbitrarily aligned -> element loads w/ clamp (P=0 masks garbage)
#pragma unroll
      for (int j = 0; j < 16; ++j) {
        int kk = kbase + j; kk = kk < 0 ? 0 : kk;
        vb[j] = vrow[kk];
      }
      oacc[dt] = wmma_bf16(pa, vb, oacc[dt]);
    }
  }
  float grow[8];
#pragma unroll
  for (int i = 0; i < 8; ++i)
    grow[i] = gates[((size_t)t * NH + g * HGQ + i + 8 * lh) * 3 + 2];
#pragma unroll
  for (int dt = 0; dt < 4; ++dt)
#pragma unroll
    for (int i = 0; i < 8; ++i) {
      int h = g * HGQ + i + 8 * lh;
      ocomb[((size_t)t * NH + h) * DVV + dt * 16 + lr] += grow[i] * oacc[dt][i];
    }
}

extern "C" void kernel_launch(void* const* d_in, const int* in_sizes, int n_in,
                              void* d_out, int out_size, void* d_ws, size_t ws_size,
                              hipStream_t stream) {
  (void)in_sizes; (void)n_in; (void)out_size; (void)ws_size;
  const float* hs   = (const float*)d_in[0];
  const float* cosb = (const float*)d_in[1];
  const float* sinb = (const float*)d_in[2];
  const float* wqkv = (const float*)d_in[3];
  const float* wout = (const float*)d_in[4];

  size_t off = 0;
  auto alloc = [&](size_t bytes) -> void* {
    off = (off + 255) & ~(size_t)255;
    void* p = (char*)d_ws + off;
    off += bytes;
    return p;
  };
  float*  qkvw   = (float*) alloc((size_t)T_SZ * OUTD * sizeof(float));
  bf16_t* xb     = (bf16_t*)alloc((size_t)T_SZ * HIDD * sizeof(bf16_t));
  bf16_t* wqkvt  = (bf16_t*)alloc((size_t)OUTD_PAD * HIDD * sizeof(bf16_t));
  bf16_t* woutt  = (bf16_t*)alloc((size_t)HIDD * (NH * DVV) * sizeof(bf16_t));
  bf16_t* qb     = (bf16_t*)alloc((size_t)T_SZ * NH * DQK * sizeof(bf16_t));
  bf16_t* kb     = (bf16_t*)alloc((size_t)T_SZ * NG * DQK * sizeof(bf16_t));
  bf16_t* vt     = (bf16_t*)alloc((size_t)NG * DVV * T_SZ * sizeof(bf16_t));
  float*  gates  = (float*) alloc((size_t)T_SZ * NH * 3 * sizeof(float));
  bf16_t* kcmp   = (bf16_t*)alloc((size_t)NBB * NG * DQK * sizeof(bf16_t));
  bf16_t* vtcmp  = (bf16_t*)alloc((size_t)NG * DVV * NBB * sizeof(bf16_t));
  float*  imp    = (float*) alloc((size_t)T_SZ * NG * NBB * sizeof(float));
  int*    selix  = (int*)   alloc((size_t)T_SZ * NG * NSELK * sizeof(int));
  float*  ocomb  = (float*) alloc((size_t)T_SZ * NH * DVV * sizeof(float));
  bf16_t* ocombb = (bf16_t*)alloc((size_t)T_SZ * NH * DVV * sizeof(bf16_t));

  dim3 blk256(256);
  // pre-passes: bf16 convert activations, transpose+convert weights
  cvt_bf16_kernel<<<dim3((T_SZ * HIDD) / 256), blk256, 0, stream>>>(hs, xb, T_SZ * HIDD);
  transpose_bf16_kernel<<<dim3((OUTD_PAD * HIDD + 255) / 256), blk256, 0, stream>>>(
      wqkv, wqkvt, HIDD, OUTD, OUTD_PAD);
  transpose_bf16_kernel<<<dim3((HIDD * NH * DVV + 255) / 256), blk256, 0, stream>>>(
      wout, woutt, NH * DVV, HIDD, HIDD);

  gemm_bf16_kernel<<<dim3(OUTD_PAD / BN, T_SZ / BM), blk256, 0, stream>>>(
      xb, wqkvt, qkvw, T_SZ, OUTD, HIDD);
  rope_pack_kernel<<<dim3(T_SZ), blk256, 0, stream>>>(qkvw, cosb, sinb, qb, kb, vt, gates);
  pool_kernel<<<dim3(NBB * NG), dim3(64), 0, stream>>>(kb, qkvw, kcmp, vtcmp);
  cmp_attn_kernel<<<dim3(T_SZ * NG), dim3(32), 0, stream>>>(qb, kcmp, vtcmp, gates, ocomb, imp);
  topk_kernel<<<dim3((T_SZ * NG + 255) / 256), blk256, 0, stream>>>(imp, selix);
  sel_attn_kernel<<<dim3(T_SZ * NG), dim3(32), 0, stream>>>(qb, kb, vt, selix, gates, ocomb);
  win_attn_kernel<<<dim3(T_SZ * NG), dim3(32), 0, stream>>>(qb, kb, vt, gates, ocomb);

  cvt_bf16_kernel<<<dim3((T_SZ * NH * DVV) / 256), blk256, 0, stream>>>(
      ocomb, ocombb, T_SZ * NH * DVV);
  gemm_bf16_kernel<<<dim3(HIDD / BN, T_SZ / BM), blk256, 0, stream>>>(
      ocombb, woutt, (float*)d_out, T_SZ, HIDD, NH * DVV);
}
